// MyModel_19851338842187
// MI455X (gfx1250) — matrix operand order repaired and checked by reading, not compiled, not run on hardware
//
#include <hip/hip_runtime.h>
#include <cstddef>
#include <cstdint>

// ---------------------------------------------------------------------------
// MI455X (gfx1250) implementation. All dense GEMMs run through
// v_wmma_f32_16x16x32_f16 (f16 A/B, f32 accum), 32x64 output tile per wave,
// with the weight panel staged into LDS via global_load_async_to_lds_b128
// (double-buffered, s_wait_asynccnt). wave32 everywhere.
// ---------------------------------------------------------------------------

typedef __attribute__((ext_vector_type(16))) _Float16 v16h;
typedef __attribute__((ext_vector_type(8)))  _Float16 v8h;
typedef __attribute__((ext_vector_type(8)))  float    v8f;

#define B_   1024
#define L_   79
#define E_   128
#define H_   8
#define D_   16
#define FF_  256
#define NL_  6
#define N_   100
#define GD_  32
#define EPS_ 1e-5f

#define MT_  2   // 16-row sub-tiles per wave
#define NT_  4   // 16-col sub-tiles per wave

__device__ __forceinline__ float wsum(float v) {
  #pragma unroll
  for (int o = 16; o > 0; o >>= 1) v += __shfl_xor(v, o, 32);
  return v;
}
__device__ __forceinline__ float wmaxr(float v) {
  #pragma unroll
  for (int o = 16; o > 0; o >>= 1) v = fmaxf(v, __shfl_xor(v, o, 32));
  return v;
}

// ---------------------------------------------------------------------------
// WMMA GEMM:  C[M,N] = A[M,Kp] * Bt[N,Kp]^T + bias, optional relu.
//   A : f16 row-major, K zero-padded to Kp (Kp % 32 == 0)
//   Bt: f16 weight pre-transposed to [N, Kp] (contiguous fragment rows)
// One wave owns a 32x64 tile = 2x4 WMMA sub-tiles (8 v_wmma per K-chunk of 32;
// A fragments reused 4x, B fragments 2x). The 64-col B panel for the current
// K-chunk (64x32 f16 = 4KB) is staged into a per-wave LDS slice with
// global_load_async_to_lds_b128, double-buffered: issue next chunk, then
// s_wait_asynccnt 8 (async loads complete in order -> current chunk resident),
// consume from LDS, repeat. Tail chunk waits asynccnt 0. No cross-wave
// sharing -> no barriers; all branches wave-uniform so EXEC stays all-1s.
// Fragment k order: contiguous k = 16*half + s in A and B (16-bit A 16x32, B 32x16, f32 C/D).
// Ragged edges: sub-tile bases clamped (duplicate compute, guarded store).
// ---------------------------------------------------------------------------
__global__ __launch_bounds__(256) void gemm_wmma_kernel(
    const _Float16* __restrict__ A, const _Float16* __restrict__ Bt,
    const float* __restrict__ bias, float* __restrict__ C,
    _Float16* __restrict__ Ch, int M, int N, int Kp, int relu)
{
  __shared__ __align__(128) _Float16 bstage[8][2][64][32];  // 64 KB / block

  int lane  = threadIdx.x & 31;
  int wslot = threadIdx.x >> 5;                  // 0..7
  long long wid = ((long long)blockIdx.x * blockDim.x + threadIdx.x) >> 5;
  int ntg = (N + 63) >> 6;                       // 64-col groups
  long long mg = wid / ntg;
  int ng = (int)(wid % ntg);
  long long m0 = mg * 32;
  if (m0 >= (long long)M) return;                // wave-uniform exit

  int half = lane >> 4;
  int l15  = lane & 15;

  // A row pointers for the 2 row sub-tiles (clamped; store-guarded)
  const _Float16* ap[MT_];
  long long mbase[MT_];
  bool mstore[MT_];
  #pragma unroll
  for (int i = 0; i < MT_; ++i) {
    long long mb = m0 + 16 * i;
    mstore[i] = (mb + 16 <= (long long)M);
    if (mb + 16 > (long long)M) mb = (long long)M - 16;
    mbase[i] = mb;
    ap[i] = A + (mb + l15) * (long long)Kp;
  }
  // column sub-tiles (clamped; store-guarded)
  int ncol[NT_];
  bool nstore[NT_];
  #pragma unroll
  for (int j = 0; j < NT_; ++j) {
    int nb = ng * 64 + 16 * j;
    nstore[j] = (nb + 16 <= N);
    if (nb + 16 > N) nb = N - 16;
    ncol[j] = nb + l15;
  }

  // async-staging lane mapping: 8 issues x (8 rows x 4 x 16B segs) = 64x32 f16
  int srow = lane >> 2;                          // 0..7
  int seg  = lane & 3;                           // 0..3 (16B each)
  int nchunks = Kp >> 5;

  // prologue: stage chunk 0 into buffer 0
  #pragma unroll
  for (int t = 0; t < 8; ++t) {
    int r = t * 8 + srow;
    int rg = ng * 64 + r; if (rg > N - 1) rg = N - 1;
    const _Float16* ga = Bt + (long long)rg * Kp + seg * 8;
    uint32_t la = (uint32_t)(uintptr_t)&bstage[wslot][0][r][seg * 8];
    asm volatile("global_load_async_to_lds_b128 %0, %1, off"
                 :: "v"(la), "v"(ga) : "memory");
  }

  v8f acc[MT_][NT_] = {};
  for (int c = 0; c < nchunks; ++c) {
    int k0 = c << 5;
    int buf = c & 1;
    if (c + 1 < nchunks) {
      int nb = buf ^ 1;
      #pragma unroll
      for (int t = 0; t < 8; ++t) {
        int r = t * 8 + srow;
        int rg = ng * 64 + r; if (rg > N - 1) rg = N - 1;
        const _Float16* ga = Bt + (long long)rg * Kp + (k0 + 32) + seg * 8;
        uint32_t la = (uint32_t)(uintptr_t)&bstage[wslot][nb][r][seg * 8];
        asm volatile("global_load_async_to_lds_b128 %0, %1, off"
                     :: "v"(la), "v"(ga) : "memory");
      }
      // in-order completion: <=8 outstanding means current chunk has landed
      asm volatile("s_wait_asynccnt 0x8" ::: "memory");
    } else {
      asm volatile("s_wait_asynccnt 0x0" ::: "memory");
    }

    // A fragments: direct global b128 loads (+ prefetch of next K tile)
    v16h af[MT_];
    #pragma unroll
    for (int i = 0; i < MT_; ++i) {
      v8h alo = *(const v8h*)(ap[i] + k0 + 16 * half);
      v8h ahi = *(const v8h*)(ap[i] + k0 + 16 * half + 8);
      af[i] = __builtin_shufflevector(alo, ahi,
                 0,1,2,3,4,5,6,7,8,9,10,11,12,13,14,15);
      if (k0 + 32 < Kp) __builtin_prefetch(ap[i] + k0 + 64, 0, 1);
    }
    // B fragments from the LDS stage (ds_load_b128 pairs)
    v16h bf[NT_];
    #pragma unroll
    for (int j = 0; j < NT_; ++j)
      bf[j] = *(const v16h*)&bstage[wslot][buf][j * 16 + l15][16 * half];

    #pragma unroll
    for (int i = 0; i < MT_; ++i)
      #pragma unroll
      for (int j = 0; j < NT_; ++j)
        acc[i][j] = __builtin_amdgcn_wmma_f32_16x16x32_f16(
            false, af[i], false, bf[j], (short)0, acc[i][j], false, false);
  }

  // epilogue
  #pragma unroll
  for (int j = 0; j < NT_; ++j) {
    if (!nstore[j]) continue;
    float bv = bias ? bias[ncol[j]] : 0.f;
    #pragma unroll
    for (int i = 0; i < MT_; ++i) {
      if (!mstore[i]) continue;
      #pragma unroll
      for (int r = 0; r < 8; ++r) {
        long long row = mbase[i] + r + 8 * half;
        float vv = acc[i][j][r] + bv;
        if (relu) vv = fmaxf(vv, 0.f);
        if (C)  C [row * (long long)N + ncol[j]] = vv;
        if (Ch) Ch[row * (long long)N + ncol[j]] = (_Float16)vv;
      }
    }
  }
}

// --------------------------- conversion kernels ----------------------------
// W[K,N] f32 -> Wt[N,Kp] f16 (transposed + K zero-pad)
__global__ void wconv_kernel(const float* __restrict__ W, _Float16* __restrict__ Wt,
                             int K, int N, int Kp) {
  long long i = (long long)blockIdx.x * blockDim.x + threadIdx.x;
  if (i >= (long long)N * Kp) return;
  int k = (int)(i % Kp);
  int n = (int)(i / Kp);
  Wt[i] = (k < K) ? (_Float16)W[(long long)k * N + n] : (_Float16)0.f;
}

// A[M,K] f32 -> Ah[M,Kp] f16 (K zero-pad)
__global__ void aconv_kernel(const float* __restrict__ A, _Float16* __restrict__ Ah,
                             long long M, int K, int Kp) {
  long long i = (long long)blockIdx.x * blockDim.x + threadIdx.x;
  if (i >= M * Kp) return;
  int k = (int)(i % Kp);
  long long m = i / Kp;
  Ah[i] = (k < K) ? (_Float16)A[m * K + k] : (_Float16)0.f;
}

// ------------------------------ transformer --------------------------------
__global__ void embed_kernel(const int* __restrict__ tok, const float* __restrict__ emb,
                             const float* __restrict__ pe,
                             float* __restrict__ x, _Float16* __restrict__ xh) {
  long long i = (long long)blockIdx.x * blockDim.x + threadIdx.x;
  if (i >= (long long)B_ * L_ * E_) return;
  int e = (int)(i % E_);
  long long bl = i / E_;
  int l = (int)(bl % L_);
  float v = emb[tok[bl] * E_ + e] + pe[l * E_ + e];
  x[i] = v;
  xh[i] = (_Float16)v;
}

// one wave per (b, head, query-row); softmax over 79 keys via shuffles
__global__ void attn_kernel(const float* __restrict__ q, const float* __restrict__ k,
                            const float* __restrict__ v, _Float16* __restrict__ of) {
  int lane = threadIdx.x & 31;
  long long wid = ((long long)blockIdx.x * blockDim.x + threadIdx.x) >> 5;
  if (wid >= (long long)B_ * H_ * L_) return;
  int lq = (int)(wid % L_);
  long long bh = wid / L_;
  int h = (int)(bh % H_);
  long long b = bh / H_;

  const float* qp = q + (b * L_ + lq) * (long long)E_ + h * D_;
  float qv[D_];
  #pragma unroll
  for (int dd = 0; dd < D_; ++dd) qv[dd] = qp[dd];

  float s[3];
  #pragma unroll
  for (int c = 0; c < 3; ++c) {
    int m = c * 32 + lane;
    float acc = -1e30f;
    if (m < L_) {
      const float* kp = k + (b * L_ + m) * (long long)E_ + h * D_;
      acc = 0.f;
      #pragma unroll
      for (int dd = 0; dd < D_; ++dd) acc += qv[dd] * kp[dd];
      acc *= 0.25f;                       // 1/sqrt(16)
    }
    s[c] = acc;
  }
  float mx = wmaxr(fmaxf(fmaxf(s[0], s[1]), s[2]));
  float p[3]; float sum = 0.f;
  #pragma unroll
  for (int c = 0; c < 3; ++c) {
    int m = c * 32 + lane;
    p[c] = (m < L_) ? expf(s[c] - mx) : 0.f;
    sum += p[c];
  }
  sum = wsum(sum);
  float inv = 1.f / sum;

  _Float16* op = of + (b * L_ + lq) * (long long)E_ + h * D_;
  #pragma unroll
  for (int dd = 0; dd < D_; ++dd) {
    float acc = 0.f;
    #pragma unroll
    for (int c = 0; c < 3; ++c) {
      int m = c * 32 + lane;
      if (m < L_) acc += p[c] * v[(b * L_ + m) * (long long)E_ + h * D_ + dd];
    }
    acc = wsum(acc);
    if (lane == 0) op[dd] = (_Float16)(acc * inv);
  }
}

// x = LN(x + r); one wave per row of 128; writes f32 + f16 copies
__global__ void add_ln_kernel(float* __restrict__ x, const float* __restrict__ r,
                              const float* __restrict__ g, const float* __restrict__ b,
                              _Float16* __restrict__ xh, long long M) {
  int lane = threadIdx.x & 31;
  long long row = ((long long)blockIdx.x * blockDim.x + threadIdx.x) >> 5;
  if (row >= M) return;
  long long off = row * E_ + lane * 4;
  float vals[4];
  #pragma unroll
  for (int j = 0; j < 4; ++j) vals[j] = x[off + j] + r[off + j];
  float s = vals[0] + vals[1] + vals[2] + vals[3];
  float mean = wsum(s) * (1.f / E_);
  float qv = 0.f;
  #pragma unroll
  for (int j = 0; j < 4; ++j) { float d = vals[j] - mean; qv += d * d; }
  float rstd = rsqrtf(wsum(qv) * (1.f / E_) + EPS_);
  #pragma unroll
  for (int j = 0; j < 4; ++j) {
    int e = lane * 4 + j;
    float y = g[e] * (vals[j] - mean) * rstd + b[e];
    x[off + j] = y;
    xh[off + j] = (_Float16)y;
  }
}

// smi = mean over L; written as f16 into comb[:, 0:128]
__global__ void mean_smi_kernel(const float* __restrict__ x, _Float16* __restrict__ comb) {
  long long i = (long long)blockIdx.x * blockDim.x + threadIdx.x;
  if (i >= (long long)B_ * E_) return;
  int e = (int)(i % E_);
  long long b = i / E_;
  float s = 0.f;
  for (int l = 0; l < L_; ++l) s += x[(b * L_ + l) * (long long)E_ + e];
  comb[b * 896 + e] = (_Float16)(s * (1.f / L_));
}

// --------------------------------- GAT -------------------------------------
__global__ void gat_e_kernel(const float* __restrict__ hG, const float* __restrict__ a,
                             float* __restrict__ e1, float* __restrict__ e2) {
  int lane = threadIdx.x & 31;
  long long w = ((long long)blockIdx.x * blockDim.x + threadIdx.x) >> 5;
  if (w >= (long long)B_ * N_) return;
  float hv = hG[w * GD_ + lane];
  float s1 = wsum(hv * a[lane]);
  float s2 = wsum(hv * a[GD_ + lane]);
  if (lane == 0) { e1[w] = s1; e2[w] = s2; }
}

__global__ void gat_att_kernel(const float* __restrict__ e1, const float* __restrict__ e2,
                               const float* __restrict__ Aadj, float* __restrict__ attG) {
  int lane = threadIdx.x & 31;
  long long w = ((long long)blockIdx.x * blockDim.x + threadIdx.x) >> 5;
  if (w >= (long long)B_ * N_) return;
  int i = (int)(w % N_);
  long long b = w / N_;
  float ei = e1[b * N_ + i];
  float vv[4];
  #pragma unroll
  for (int c = 0; c < 4; ++c) {
    int j = c * 32 + lane;
    float val = -9e15f;
    if (j < N_) {
      float e = ei + e2[b * N_ + j];
      e = (e > 0.f) ? e : 0.2f * e;               // leaky_relu 0.2
      val = (Aadj[(b * N_ + i) * (long long)N_ + j] > 0.f) ? e : -9e15f;
    }
    vv[c] = val;
  }
  float mx = wmaxr(fmaxf(fmaxf(vv[0], vv[1]), fmaxf(vv[2], vv[3])));
  float p[4]; float sum = 0.f;
  #pragma unroll
  for (int c = 0; c < 4; ++c) {
    int j = c * 32 + lane;
    p[c] = (j < N_) ? expf(vv[c] - mx) : 0.f;
    sum += p[c];
  }
  sum = wsum(sum);
  float inv = 1.f / sum;
  #pragma unroll
  for (int c = 0; c < 4; ++c) {
    int j = c * 32 + lane;
    if (j < N_) attG[(b * N_ + i) * (long long)N_ + j] = p[c] * inv;
  }
}

// go = elu(att @ h), replicated x4 heads into goh[B, 12800] f16
__global__ void gat_go_kernel(const float* __restrict__ attG, const float* __restrict__ hG,
                              _Float16* __restrict__ goh) {
  int lane = threadIdx.x & 31;                    // lane == GD index (GD=32)
  long long w = ((long long)blockIdx.x * blockDim.x + threadIdx.x) >> 5;
  if (w >= (long long)B_ * N_) return;
  int i = (int)(w % N_);
  long long b = w / N_;
  const float* ar = attG + (b * N_ + i) * (long long)N_;
  float acc = 0.f;
  for (int j = 0; j < N_; ++j) acc += ar[j] * hG[(b * N_ + j) * (long long)GD_ + lane];
  float ev = (acc > 0.f) ? acc : (expf(acc) - 1.f);  // elu
  _Float16 hv = (_Float16)ev;
  long long base = b * 12800 + (long long)i * 128 + lane;
  #pragma unroll
  for (int rep = 0; rep < 4; ++rep) goh[base + rep * 32] = hv;
}

// ------------------------------ BatchNorm ----------------------------------
__global__ void bn_stats_kernel(const float* __restrict__ X, float* __restrict__ mean,
                                float* __restrict__ var, int Bn, int F) {
  int f = blockIdx.x;
  int tid = threadIdx.x;
  __shared__ float ss[256], sq[256];
  float s = 0.f, q = 0.f;
  for (int r = tid; r < Bn; r += blockDim.x) {
    float v = X[(long long)r * F + f];
    s += v; q += v * v;
  }
  ss[tid] = s; sq[tid] = q;
  __syncthreads();
  for (int st = 128; st > 0; st >>= 1) {
    if (tid < st) { ss[tid] += ss[tid + st]; sq[tid] += sq[tid + st]; }
    __syncthreads();
  }
  if (tid == 0) {
    float m = ss[0] / Bn;
    mean[f] = m;
    var[f] = sq[0] / Bn - m * m;                 // biased variance
  }
}

// y = relu(g*(x-m)*rsqrt(v+eps)+b) stored as f16 at dst[r*stride + off + f]
__global__ void bn_apply_kernel(const float* __restrict__ X, const float* __restrict__ g,
                                const float* __restrict__ b, const float* __restrict__ mean,
                                const float* __restrict__ var, _Float16* __restrict__ dst,
                                int F, int dstStride, int dstOff) {
  long long i = (long long)blockIdx.x * blockDim.x + threadIdx.x;
  if (i >= (long long)B_ * F) return;
  int f = (int)(i % F);
  long long r = i / F;
  float y = g[f] * (X[i] - mean[f]) * rsqrtf(var[f] + EPS_) + b[f];
  y = fmaxf(y, 0.f);
  dst[r * dstStride + dstOff + f] = (_Float16)y;
}

// ------------------------------- head --------------------------------------
__global__ void zero_kernel(float* p) {
  if (blockIdx.x == 0 && threadIdx.x == 0) p[0] = 0.f;
}

__global__ void head_kernel(const _Float16* __restrict__ zh, const float* __restrict__ fcW,
                            const float* __restrict__ fcb, const float* __restrict__ label,
                            float* __restrict__ out) {
  int lane = threadIdx.x & 31;
  long long w = ((long long)blockIdx.x * blockDim.x + threadIdx.x) >> 5;
  if (w >= (long long)B_) return;
  float acc = 0.f;
  #pragma unroll
  for (int c = 0; c < 4; ++c) {
    int e = c * 32 + lane;
    acc += (float)zh[w * 128 + e] * fcW[e];
  }
  acc = wsum(acc);
  if (lane == 0) {
    float logit = acc + fcb[0];
    float prob = 1.f / (1.f + expf(-logit));
    out[w] = prob;
    float pc = fminf(fmaxf(prob, 1e-7f), 1.f - 1e-7f);
    float lb = label[w];
    float bce = -(lb * logf(pc) + (1.f - lb) * logf(1.f - pc));
    atomicAdd(out + B_, bce * (1.f / (float)B_));
  }
}

// ---------------------------------------------------------------------------
// host-side launch helpers
// ---------------------------------------------------------------------------
static void launch_gemm(hipStream_t s, const _Float16* A, const _Float16* Bt,
                        const float* bias, float* C, _Float16* Ch,
                        int M, int N, int Kp, int relu) {
  long long waves = (long long)((M + 31) / 32) * ((N + 63) / 64);
  long long thr = waves * 32;
  long long grid = (thr + 255) / 256;
  gemm_wmma_kernel<<<dim3((unsigned)grid), dim3(256), 0, s>>>(A, Bt, bias, C, Ch, M, N, Kp, relu);
}
static void launch_wconv(hipStream_t s, const float* W, _Float16* Wt, int K, int N, int Kp) {
  long long total = (long long)N * Kp;
  wconv_kernel<<<dim3((unsigned)((total + 255) / 256)), dim3(256), 0, s>>>(W, Wt, K, N, Kp);
}
static void launch_aconv(hipStream_t s, const float* A, _Float16* Ah, long long M, int K, int Kp) {
  long long total = M * Kp;
  aconv_kernel<<<dim3((unsigned)((total + 255) / 256)), dim3(256), 0, s>>>(A, Ah, M, K, Kp);
}
static void launch_bn(hipStream_t s, const float* X, const float* g, const float* b,
                      float* mean, float* var, _Float16* dst, int F, int stride, int off) {
  bn_stats_kernel<<<dim3(F), dim3(256), 0, s>>>(X, mean, var, B_, F);
  long long total = (long long)B_ * F;
  bn_apply_kernel<<<dim3((unsigned)((total + 255) / 256)), dim3(256), 0, s>>>(
      X, g, b, mean, var, dst, F, stride, off);
}

extern "C" void kernel_launch(void* const* d_in, const int* in_sizes, int n_in,
                              void* d_out, int out_size, void* d_ws, size_t ws_size,
                              hipStream_t stream) {
  (void)in_sizes; (void)n_in; (void)out_size; (void)ws_size;

  // ---- inputs (setup_inputs dict order; params leaves in declaration order)
  const int*   tok   = (const int*)d_in[0];
  const float* fin[5]= {(const float*)d_in[1], (const float*)d_in[2],
                        (const float*)d_in[3], (const float*)d_in[4],
                        (const float*)d_in[5]};
  const float* Xg    = (const float*)d_in[6];
  const float* Aadj  = (const float*)d_in[7];
  const float* label = (const float*)d_in[8];
  const float* emb   = (const float*)d_in[9];
  const float* pe    = (const float*)d_in[10];
  const float* Wq = (const float*)d_in[11]; const float* bq = (const float*)d_in[12];
  const float* Wk = (const float*)d_in[13]; const float* bk = (const float*)d_in[14];
  const float* Wv = (const float*)d_in[15]; const float* bv = (const float*)d_in[16];
  const float* Wo = (const float*)d_in[17]; const float* bo = (const float*)d_in[18];
  const float* ln1g = (const float*)d_in[19]; const float* ln1b = (const float*)d_in[20];
  const float* ln2g = (const float*)d_in[21]; const float* ln2b = (const float*)d_in[22];
  const float* W1 = (const float*)d_in[23]; const float* b1 = (const float*)d_in[24];
  const float* W2 = (const float*)d_in[25]; const float* b2 = (const float*)d_in[26];
  const float* gatW = (const float*)d_in[27]; const float* gatA = (const float*)d_in[28];
  const float* gW1 = (const float*)d_in[29]; const float* gb1 = (const float*)d_in[30];
  const float* gg1 = (const float*)d_in[31]; const float* gbe1 = (const float*)d_in[32];
  const float* gW2 = (const float*)d_in[33]; const float* gb2 = (const float*)d_in[34];
  const float* gg2 = (const float*)d_in[35]; const float* gbe2 = (const float*)d_in[36];
  const float* pW = (const float*)d_in[37]; const float* pb = (const float*)d_in[38];
  const float* pg = (const float*)d_in[39]; const float* pbe = (const float*)d_in[40];
  const float* fcW = (const float*)d_in[41]; const float* fcb = (const float*)d_in[42];
  auto fpP = [&](int br, int ly, int t) -> const float* {
    return (const float*)d_in[43 + (br * 3 + ly) * 4 + t];
  };

  const int fpK [5] = {1024, 881, 1024, 780, 4860};
  const int fpKp[5] = {1024, 896, 1024, 800, 4864};

  // ---- workspace bump allocator -------------------------------------------
  char* wsb = (char*)d_ws;
  size_t cur = 0;
  auto alloc = [&](size_t bytes) -> void* {
    void* p = wsb + cur;
    cur = (cur + bytes + 255) & ~(size_t)255;
    return p;
  };

  // persistent: converted f16 weights + concat buffer + BN stats
  _Float16* wqt = (_Float16*)alloc((size_t)6 * 128 * 128 * 2);
  _Float16* wkt = (_Float16*)alloc((size_t)6 * 128 * 128 * 2);
  _Float16* wvt = (_Float16*)alloc((size_t)6 * 128 * 128 * 2);
  _Float16* wot = (_Float16*)alloc((size_t)6 * 128 * 128 * 2);
  _Float16* w1t = (_Float16*)alloc((size_t)6 * 256 * 128 * 2);
  _Float16* w2t = (_Float16*)alloc((size_t)6 * 128 * 256 * 2);
  _Float16* gatWt = (_Float16*)alloc((size_t)32 * 96 * 2);
  _Float16* gW1t = (_Float16*)alloc((size_t)1024 * 12800 * 2);
  _Float16* gW2t = (_Float16*)alloc((size_t)128 * 1024 * 2);
  _Float16* fpWt[5][3];
  for (int i = 0; i < 5; ++i) {
    fpWt[i][0] = (_Float16*)alloc((size_t)512 * fpKp[i] * 2);
    fpWt[i][1] = (_Float16*)alloc((size_t)256 * 512 * 2);
    fpWt[i][2] = (_Float16*)alloc((size_t)128 * 256 * 2);
  }
  _Float16* pWt = (_Float16*)alloc((size_t)128 * 896 * 2);
  float* bn_mean = (float*)alloc(1024 * 4);
  float* bn_var  = (float*)alloc(1024 * 4);
  _Float16* comb = (_Float16*)alloc((size_t)B_ * 896 * 2);
  size_t scratch0 = cur;

  // ---- weight conversions (fp32 -> transposed, padded f16) ----------------
  for (int l = 0; l < 6; ++l) {
    launch_wconv(stream, Wq + (size_t)l * 16384, wqt + (size_t)l * 16384, 128, 128, 128);
    launch_wconv(stream, Wk + (size_t)l * 16384, wkt + (size_t)l * 16384, 128, 128, 128);
    launch_wconv(stream, Wv + (size_t)l * 16384, wvt + (size_t)l * 16384, 128, 128, 128);
    launch_wconv(stream, Wo + (size_t)l * 16384, wot + (size_t)l * 16384, 128, 128, 128);
    launch_wconv(stream, W1 + (size_t)l * 32768, w1t + (size_t)l * 32768, 128, 256, 128);
    launch_wconv(stream, W2 + (size_t)l * 32768, w2t + (size_t)l * 32768, 256, 128, 256);
  }
  launch_wconv(stream, gatW, gatWt, 75, 32, 96);
  launch_wconv(stream, gW1, gW1t, 12800, 1024, 12800);
  launch_wconv(stream, gW2, gW2t, 1024, 128, 1024);
  for (int i = 0; i < 5; ++i) {
    launch_wconv(stream, fpP(i, 0, 0), fpWt[i][0], fpK[i], 512, fpKp[i]);
    launch_wconv(stream, fpP(i, 1, 0), fpWt[i][1], 512, 256, 512);
    launch_wconv(stream, fpP(i, 2, 0), fpWt[i][2], 256, 128, 256);
  }
  launch_wconv(stream, pW, pWt, 896, 128, 896);

  // ======================= phase 1: SMILES transformer =====================
  cur = scratch0;
  const long long M = (long long)B_ * L_;               // 80896 rows
  float*    x  = (float*)   alloc((size_t)M * 128 * 4);
  _Float16* xh = (_Float16*)alloc((size_t)M * 128 * 2);
  float*    qb = (float*)   alloc((size_t)M * 128 * 4);
  float*    kb = (float*)   alloc((size_t)M * 128 * 4);
  float*    vb = (float*)   alloc((size_t)M * 128 * 4);
  _Float16* of = (_Float16*)alloc((size_t)M * 128 * 2);
  float*    t0  = qb;             // o-proj output (q dead after attention)
  _Float16* h1h = (_Float16*)kb;  // FFN hidden f16 [M,256] (k dead)
  float*    h2  = vb;             // FFN output    (v dead)

  embed_kernel<<<dim3((unsigned)((M * 128 + 255) / 256)), dim3(256), 0, stream>>>(
      tok, emb, pe, x, xh);

  for (int l = 0; l < NL_; ++l) {
    const size_t wo = (size_t)l * 16384;
    launch_gemm(stream, xh, wqt + wo, bq + l * 128, qb, nullptr, (int)M, 128, 128, 0);
    launch_gemm(stream, xh, wkt + wo, bk + l * 128, kb, nullptr, (int)M, 128, 128, 0);
    launch_gemm(stream, xh, wvt + wo, bv + l * 128, vb, nullptr, (int)M, 128, 128, 0);
    {
      long long waves = (long long)B_ * H_ * L_;
      attn_kernel<<<dim3((unsigned)((waves * 32 + 255) / 256)), dim3(256), 0, stream>>>(
          qb, kb, vb, of);
    }
    launch_gemm(stream, of, wot + wo, bo + l * 128, t0, nullptr, (int)M, 128, 128, 0);
    add_ln_kernel<<<dim3((unsigned)((M * 32 + 255) / 256)), dim3(256), 0, stream>>>(
        x, t0, ln1g + l * 128, ln1b + l * 128, xh, M);
    launch_gemm(stream, xh, w1t + (size_t)l * 32768, b1 + l * 256,
                nullptr, h1h, (int)M, 256, 128, 1 /*relu*/);
    launch_gemm(stream, h1h, w2t + (size_t)l * 32768, b2 + l * 128,
                h2, nullptr, (int)M, 128, 256, 0);
    add_ln_kernel<<<dim3((unsigned)((M * 32 + 255) / 256)), dim3(256), 0, stream>>>(
        x, h2, ln2g + l * 128, ln2b + l * 128, xh, M);
  }
  mean_smi_kernel<<<dim3((unsigned)((B_ * E_ + 255) / 256)), dim3(256), 0, stream>>>(x, comb);

  // ======================= phase 2: GAT + FP branches ======================
  cur = scratch0;   // transformer scratch dead; re-use the pool
  _Float16* Xh   = (_Float16*)alloc((size_t)102400 * 96 * 2);
  float*    hG   = (float*)   alloc((size_t)102400 * 32 * 4);
  float*    e1   = (float*)   alloc((size_t)102400 * 4);
  float*    e2   = (float*)   alloc((size_t)102400 * 4);
  float*    attG = (float*)   alloc((size_t)B_ * N_ * N_ * 4);
  _Float16* goh  = (_Float16*)alloc((size_t)B_ * 12800 * 2);
  float*    g1   = (float*)   alloc((size_t)B_ * 1024 * 4);
  _Float16* g1h  = (_Float16*)alloc((size_t)B_ * 1024 * 2);
  float*    g2   = (float*)   alloc((size_t)B_ * 128 * 4);
  _Float16* finh = (_Float16*)alloc((size_t)B_ * 4864 * 2);
  float*    s1   = (float*)   alloc((size_t)B_ * 512 * 4);
  _Float16* s1h  = (_Float16*)alloc((size_t)B_ * 512 * 2);
  float*    s2   = (float*)   alloc((size_t)B_ * 256 * 4);
  _Float16* s2h  = (_Float16*)alloc((size_t)B_ * 256 * 2);
  float*    s3   = (float*)   alloc((size_t)B_ * 128 * 4);
  float*    zf   = (float*)   alloc((size_t)B_ * 128 * 4);
  _Float16* zh   = (_Float16*)alloc((size_t)B_ * 128 * 2);

  // GAT
  launch_aconv(stream, Xg, Xh, 102400, 75, 96);
  launch_gemm(stream, Xh, gatWt, nullptr, hG, nullptr, 102400, 32, 96, 0);
  {
    long long waves = (long long)B_ * N_;
    unsigned blk = (unsigned)((waves * 32 + 255) / 256);
    gat_e_kernel  <<<dim3(blk), dim3(256), 0, stream>>>(hG, gatA, e1, e2);
    gat_att_kernel<<<dim3(blk), dim3(256), 0, stream>>>(e1, e2, Aadj, attG);
    gat_go_kernel <<<dim3(blk), dim3(256), 0, stream>>>(attG, hG, goh);
  }
  launch_gemm(stream, goh, gW1t, gb1, g1, nullptr, B_, 1024, 12800, 0);
  launch_bn(stream, g1, gg1, gbe1, bn_mean, bn_var, g1h, 1024, 1024, 0);
  launch_gemm(stream, g1h, gW2t, gb2, g2, nullptr, B_, 128, 1024, 0);
  launch_bn(stream, g2, gg2, gbe2, bn_mean, bn_var, comb, 128, 896, 128);

  // fingerprint MLPs
  for (int i = 0; i < 5; ++i) {
    launch_aconv(stream, fin[i], finh, B_, fpK[i], fpKp[i]);
    launch_gemm(stream, finh, fpWt[i][0], fpP(i, 0, 1), s1, nullptr, B_, 512, fpKp[i], 0);
    launch_bn(stream, s1, fpP(i, 0, 2), fpP(i, 0, 3), bn_mean, bn_var, s1h, 512, 512, 0);
    launch_gemm(stream, s1h, fpWt[i][1], fpP(i, 1, 1), s2, nullptr, B_, 256, 512, 0);
    launch_bn(stream, s2, fpP(i, 1, 2), fpP(i, 1, 3), bn_mean, bn_var, s2h, 256, 256, 0);
    launch_gemm(stream, s2h, fpWt[i][2], fpP(i, 2, 1), s3, nullptr, B_, 128, 256, 0);
    launch_bn(stream, s3, fpP(i, 2, 2), fpP(i, 2, 3), bn_mean, bn_var,
              comb, 128, 896, 256 + i * 128);
  }

  // fusion head
  launch_gemm(stream, comb, pWt, pb, zf, nullptr, B_, 128, 896, 0);
  launch_bn(stream, zf, pg, pbe, bn_mean, bn_var, zh, 128, 128, 0);
  zero_kernel<<<dim3(1), dim3(1), 0, stream>>>((float*)d_out + B_);
  head_kernel<<<dim3((B_ * 32) / 256), dim3(256), 0, stream>>>(
      zh, fcW, fcb, label, (float*)d_out);
}
